// LSTM_Encoder_26259430048120
// MI455X (gfx1250) — matrix operand order repaired and checked by reading, not compiled, not run on hardware
//
#include <hip/hip_runtime.h>
#include <hip/hip_bf16.h>
#include <math.h>

// ---------------- problem constants ----------------
#define BB 16          // batch
#define TT 128         // seq len
#define D_IN 1268
#define KP0 1280       // D_IN padded to mult of 32
#define LSTMD 800
#define HH 400         // hidden per direction
#define KP1 832        // LSTMD padded to mult of 32
#define KPH 416        // HH padded to mult of 32
#define G4H 1600       // 4*HH
#define NP 8001        // (T-1 choose 2) with T-1=127

typedef __bf16 bf16_t;
typedef bf16_t v16bf __attribute__((ext_vector_type(16)));
typedef bf16_t v8bf  __attribute__((ext_vector_type(8)));
typedef float  v8f   __attribute__((ext_vector_type(8)));

__device__ __forceinline__ v16bf load_frag(const bf16_t* p0, const bf16_t* p1) {
    v8bf lo = *(const v8bf*)p0;
    v8bf hi = *(const v8bf*)p1;
    v16bf r;
#pragma unroll
    for (int i = 0; i < 8; ++i) { r[i] = lo[i]; r[i + 8] = hi[i]; }
    return r;
}

__device__ __forceinline__ float sigmoidf_(float x) {
    return 1.0f / (1.0f + __expf(-x));
}
__device__ __forceinline__ float tanhf_(float x) {
    // tanh(x) = 2*sigmoid(2x) - 1 (VALU-only transcendental path)
    return 2.0f * sigmoidf_(2.0f * x) - 1.0f;
}

// ---------------- kernel 1: embedding gather -> packed bf16 X (T*B rows, KP0 cols) ----------------
__global__ void embed_pack_kernel(const int* __restrict__ word,
                                  const int* __restrict__ word_idxs,
                                  const int* __restrict__ pos_idxs,
                                  const int* __restrict__ dep_idxs,
                                  const int* __restrict__ ent_idxs,
                                  const int* __restrict__ ent_iob_idxs,
                                  const float* __restrict__ bert_t,
                                  const float* __restrict__ word_t,
                                  const float* __restrict__ pos_t,
                                  const float* __restrict__ dep_t,
                                  const float* __restrict__ ent_t,
                                  const float* __restrict__ eiob_t,
                                  bf16_t* __restrict__ xb) {
    int r = blockIdx.x;              // row = t*B + b
    int t = r >> 4, b = r & 15;
    int s = b * TT + t;              // (B,T) source index
    int w  = word[s];
    int wi = word_idxs[s]; if (wi >= 50000) wi = 1;  // UNK
    int pi = pos_idxs[s];
    int di = dep_idxs[s];
    int ei = ent_idxs[s];
    int bi = ent_iob_idxs[s];
    bf16_t* dst = xb + (size_t)r * KP0;
    for (int c = threadIdx.x; c < KP0; c += blockDim.x) {
        float v;
        if      (c < 768)  v = bert_t[(size_t)w * 768 + c];
        else if (c < 1068) v = word_t[(size_t)wi * 300 + (c - 768)];
        else if (c < 1118) v = pos_t[pi * 50 + (c - 1068)];
        else if (c < 1168) v = dep_t[di * 50 + (c - 1118)];
        else if (c < 1218) v = ent_t[ei * 50 + (c - 1168)];
        else if (c < 1268) v = eiob_t[bi * 50 + (c - 1218)];
        else               v = 0.0f;
        dst[c] = (bf16_t)v;
    }
}

// ---------------- kernel 2: weight fp32 -> bf16, pad K to Kp ----------------
__global__ void pack_w_kernel(const float* __restrict__ src, bf16_t* __restrict__ dst,
                              int rows, int K, int Kp) {
    int i = blockIdx.x * blockDim.x + threadIdx.x;
    int total = rows * Kp;
    if (i >= total) return;
    int r = i / Kp, c = i - r * Kp;
    dst[i] = (c < K) ? (bf16_t)src[(size_t)r * K + c] : (bf16_t)0.0f;
}

__global__ void zero_bf16_kernel(bf16_t* __restrict__ p, int n) {
    int i = blockIdx.x * blockDim.x + threadIdx.x;
    if (i < n) p[i] = (bf16_t)0.0f;
}

// ---------------- kernel 3: Zin = X @ W^T + bias  (M=2048, N=1600, K=Kp) ----------------
// grid (25, 16), block 256 (8 waves); each wave: one M-tile x FOUR N-tiles,
// one A fragment reused across 4 WMMAs per k-step (4x less A traffic).
#define NT_PER_WAVE 4
__global__ void gemm_zin_kernel(const bf16_t* __restrict__ X,
                                const bf16_t* __restrict__ W,
                                const float*  __restrict__ bias,
                                float* __restrict__ Z, int Kp) {
    int lane = threadIdx.x & 31;
    int wave = threadIdx.x >> 5;
    int nTile0 = blockIdx.x * NT_PER_WAVE;  // 0,4,...,96
    int mTile  = blockIdx.y * 8 + wave;     // 0..127
    int ln   = lane & 15;                   // A row m == B col n == C col n
    int half = lane >> 4;

    const bf16_t* arow = X + (size_t)(mTile * 16 + ln) * Kp;
    const bf16_t* brow0 = W + (size_t)(nTile0 * 16 + ln) * Kp;  // tile j offset: j*16*Kp

    v8f acc[NT_PER_WAVE];
#pragma unroll
    for (int j = 0; j < NT_PER_WAVE; ++j)
        acc[j] = (v8f){0.f, 0.f, 0.f, 0.f, 0.f, 0.f, 0.f, 0.f};

    int nk = Kp >> 5;
    for (int kb = 0; kb < nk; ++kb) {
        int ka = kb * 32 + half * 8;        // A: split 8+8 (K, K+16)
        v16bf a = load_frag(arow + ka, arow + ka + 16);
        int kbb = kb * 32 + half * 8;       // B: split 8+8 (K, K+16)
#pragma unroll
        for (int j = 0; j < NT_PER_WAVE; ++j) {
            const bf16_t* bp = brow0 + (size_t)j * 16 * Kp + kbb;
            v16bf b = load_frag(bp, bp + 16);
            acc[j] = __builtin_amdgcn_wmma_f32_16x16x32_bf16(false, a, false, b,
                                                             (short)0, acc[j], false, false);
        }
    }
#pragma unroll
    for (int j = 0; j < NT_PER_WAVE; ++j) {
        int nTile = nTile0 + j;
        float bv = bias[nTile * 16 + ln];
#pragma unroll
        for (int r = 0; r < 8; ++r) {
            int mm = r + half * 8;          // C/D row
            Z[(size_t)(mTile * 16 + mm) * G4H + nTile * 16 + ln] = acc[j][r] + bv;
        }
    }
}

// ---------------- kernel 4: sequential LSTM recurrence (one WG per direction) ----------------
// block = 800 threads = 25 waves; wave w owns unit-tile u=w (16 hidden units, all 4 gates).
__global__ __launch_bounds__(800, 1)
void lstm_rec_kernel(const float* __restrict__ zin_f, const float* __restrict__ zin_b,
                     const bf16_t* __restrict__ whh_f, const bf16_t* __restrict__ whh_b,
                     bf16_t* __restrict__ x2b,          // next-layer input (if !is_final)
                     float* __restrict__ hf_out,        // (B,T,HH) if is_final
                     float* __restrict__ hb_out,
                     int is_final) {
    int dir = blockIdx.x;
    const float*  zin = dir ? zin_b : zin_f;
    const bf16_t* whh = dir ? whh_b : whh_f;

    __shared__ __align__(16) bf16_t h_lds[16][KPH];
    __shared__ __align__(16) float  c_lds[16][HH];

    for (int i = threadIdx.x; i < 16 * KPH; i += blockDim.x)
        (&h_lds[0][0])[i] = (bf16_t)0.0f;
    for (int i = threadIdx.x; i < 16 * HH; i += blockDim.x)
        (&c_lds[0][0])[i] = 0.0f;
    __syncthreads();

    int lane = threadIdx.x & 31;
    int ut   = threadIdx.x >> 5;    // unit tile 0..24
    int ln   = lane & 15;
    int half = lane >> 4;
    int ug   = ut * 16 + ln;        // hidden unit index for this lane's column

    for (int step = 0; step < TT; ++step) {
        int t = dir ? (TT - 1 - step) : step;

        v8f acc[4];
#pragma unroll
        for (int g = 0; g < 4; ++g)
            acc[g] = (v8f){0.f, 0.f, 0.f, 0.f, 0.f, 0.f, 0.f, 0.f};

        // z_r = h_prev @ Whh^T for the 4 gate tiles of this unit tile
        for (int kb = 0; kb < (KPH >> 5); ++kb) {
            int ka = kb * 32 + half * 8;
            v16bf a = load_frag(&h_lds[ln][ka], &h_lds[ln][ka + 16]);
#pragma unroll
            for (int g = 0; g < 4; ++g) {
                const bf16_t* brow = whh + (size_t)(g * HH + ut * 16 + ln) * KPH
                                         + kb * 32 + half * 8;
                __builtin_prefetch(brow + KPH, 0, 0);
                v16bf b = load_frag(brow, brow + 16);
                acc[g] = __builtin_amdgcn_wmma_f32_16x16x32_bf16(false, a, false, b,
                                                                 (short)0, acc[g], false, false);
            }
        }
        __syncthreads();   // all reads of h_lds complete before updates

        const float* zrow_base = zin + (size_t)(t * 16) * G4H;
#pragma unroll
        for (int r = 0; r < 8; ++r) {
            int b = r + half * 8;   // batch row
            const float* zr = zrow_base + (size_t)b * G4H;
            float zi = acc[0][r] + zr[0 * HH + ug];
            float zf = acc[1][r] + zr[1 * HH + ug];
            float zg = acc[2][r] + zr[2 * HH + ug];
            float zo = acc[3][r] + zr[3 * HH + ug];
            float cp = c_lds[b][ug];
            float cn = sigmoidf_(zf) * cp + sigmoidf_(zi) * tanhf_(zg);
            float hn = sigmoidf_(zo) * tanhf_(cn);
            c_lds[b][ug] = cn;
            h_lds[b][ug] = (bf16_t)hn;
            if (is_final) {
                float* hout = dir ? hb_out : hf_out;
                hout[(size_t)(b * TT + t) * HH + ug] = hn;
            } else {
                x2b[(size_t)(t * 16 + b) * KP1 + dir * HH + ug] = (bf16_t)hn;
            }
        }
        __syncthreads();   // new h visible before next step's reads
    }
}

// ---------------- kernel 5: span differences ----------------
// out[b][p][0:400]  = hf[b][c] - hf[b][a]
// out[b][p][400:800]= hb[b][a+1] - hb[b][c+1]
__global__ void span_kernel(const float* __restrict__ hf, const float* __restrict__ hb,
                            float* __restrict__ out) {
    int idx = blockIdx.x;                // b*NP + p
    int b = idx / NP;
    int p = idx - b * NP;
    // decode p -> (a,c): cum(a) = a*(253-a)/2, a in [0,125], c in [a+1,126]
    int a = (int)((253.0f - sqrtf(253.0f * 253.0f - 8.0f * (float)p)) * 0.5f);
    if (a > 125) a = 125;
    if (a < 0) a = 0;
    while (a > 0 && (a * (253 - a)) / 2 > p) --a;
    while (a < 125 && ((a + 1) * (253 - (a + 1))) / 2 <= p) ++a;
    int c = p - (a * (253 - a)) / 2 + a + 1;

    const float* hfb = hf + (size_t)b * TT * HH;
    const float* hbb = hb + (size_t)b * TT * HH;
    float* dst = out + (size_t)idx * (2 * HH);
    for (int e = threadIdx.x; e < 2 * HH; e += blockDim.x) {
        float v;
        if (e < HH) v = hfb[(size_t)c * HH + e] - hfb[(size_t)a * HH + e];
        else {
            int u = e - HH;
            v = hbb[(size_t)(a + 1) * HH + u] - hbb[(size_t)(c + 1) * HH + u];
        }
        dst[e] = v;
    }
}

// ---------------- kernel 6: sentence lengths ----------------
__global__ void senlens_kernel(const int* __restrict__ pos_idxs, float* __restrict__ out) {
    __shared__ int cnt;
    int b = blockIdx.x;
    if (threadIdx.x == 0) cnt = 0;
    __syncthreads();
    int c = 0;
    for (int t = threadIdx.x; t < TT; t += blockDim.x)
        if (pos_idxs[b * TT + t] != 0) ++c;
    atomicAdd(&cnt, c);
    __syncthreads();
    if (threadIdx.x == 0) out[b] = (float)(cnt - 2);
}

// ---------------- host launcher ----------------
static inline size_t align256(size_t x) { return (x + 255) & ~(size_t)255; }

extern "C" void kernel_launch(void* const* d_in, const int* in_sizes, int n_in,
                              void* d_out, int out_size, void* d_ws, size_t ws_size,
                              hipStream_t stream) {
    (void)in_sizes; (void)n_in; (void)out_size; (void)ws_size;

    const int* word     = (const int*)d_in[0];
    const int* widx     = (const int*)d_in[1];
    const int* pidx     = (const int*)d_in[2];
    const int* didx     = (const int*)d_in[3];
    const int* eidx     = (const int*)d_in[4];
    const int* eiidx    = (const int*)d_in[5];
    const float* bert_t = (const float*)d_in[6];
    const float* word_t = (const float*)d_in[7];
    const float* pos_t  = (const float*)d_in[8];
    const float* dep_t  = (const float*)d_in[9];
    const float* ent_t  = (const float*)d_in[10];
    const float* eiob_t = (const float*)d_in[11];
    // lstm_params flatten: layer0: (Wih_f,Whh_f,b_f),(Wih_b,Whh_b,b_b); layer1 likewise
    const float* wih0f = (const float*)d_in[12];
    const float* whh0f = (const float*)d_in[13];
    const float* b0f   = (const float*)d_in[14];
    const float* wih0b = (const float*)d_in[15];
    const float* whh0b = (const float*)d_in[16];
    const float* b0b   = (const float*)d_in[17];
    const float* wih1f = (const float*)d_in[18];
    const float* whh1f = (const float*)d_in[19];
    const float* b1f   = (const float*)d_in[20];
    const float* wih1b = (const float*)d_in[21];
    const float* whh1b = (const float*)d_in[22];
    const float* b1b   = (const float*)d_in[23];

    // workspace carve-up
    char* ws = (char*)d_ws;
    size_t off = 0;
    auto carve = [&](size_t bytes) { char* p = ws + off; off = align256(off + bytes); return p; };

    bf16_t* xb    = (bf16_t*)carve((size_t)TT * BB * KP0 * 2);
    bf16_t* x2b   = (bf16_t*)carve((size_t)TT * BB * KP1 * 2);
    float*  zin_f = (float*) carve((size_t)TT * BB * G4H * 4);
    float*  zin_b = (float*) carve((size_t)TT * BB * G4H * 4);
    float*  hf_o  = (float*) carve((size_t)BB * TT * HH * 4);
    float*  hb_o  = (float*) carve((size_t)BB * TT * HH * 4);
    bf16_t* pw0f  = (bf16_t*)carve((size_t)G4H * KP0 * 2);
    bf16_t* pw0b  = (bf16_t*)carve((size_t)G4H * KP0 * 2);
    bf16_t* pw1f  = (bf16_t*)carve((size_t)G4H * KP1 * 2);
    bf16_t* pw1b  = (bf16_t*)carve((size_t)G4H * KP1 * 2);
    bf16_t* ph0f  = (bf16_t*)carve((size_t)G4H * KPH * 2);
    bf16_t* ph0b  = (bf16_t*)carve((size_t)G4H * KPH * 2);
    bf16_t* ph1f  = (bf16_t*)carve((size_t)G4H * KPH * 2);
    bf16_t* ph1b  = (bf16_t*)carve((size_t)G4H * KPH * 2);

    // --- pack weights to bf16 (padded K) ---
    auto packw = [&](const float* s, bf16_t* d, int K, int Kp) {
        int total = G4H * Kp;
        pack_w_kernel<<<(total + 255) / 256, 256, 0, stream>>>(s, d, G4H, K, Kp);
    };
    packw(wih0f, pw0f, D_IN, KP0);
    packw(wih0b, pw0b, D_IN, KP0);
    packw(wih1f, pw1f, LSTMD, KP1);
    packw(wih1b, pw1b, LSTMD, KP1);
    packw(whh0f, ph0f, HH, KPH);
    packw(whh0b, ph0b, HH, KPH);
    packw(whh1f, ph1f, HH, KPH);
    packw(whh1b, ph1b, HH, KPH);

    // --- embeddings ---
    embed_pack_kernel<<<TT * BB, 256, 0, stream>>>(word, widx, pidx, didx, eidx, eiidx,
                                                   bert_t, word_t, pos_t, dep_t, ent_t, eiob_t, xb);
    {   // zero next-layer input (incl. K padding)
        int n = TT * BB * KP1;
        zero_bf16_kernel<<<(n + 255) / 256, 256, 0, stream>>>(x2b, n);
    }

    dim3 ggrid(25, 16);   // 25 groups of 4 N-tiles, 16*8 waves of M-tiles
    // --- layer 0 ---
    gemm_zin_kernel<<<ggrid, 256, 0, stream>>>(xb, pw0f, b0f, zin_f, KP0);
    gemm_zin_kernel<<<ggrid, 256, 0, stream>>>(xb, pw0b, b0b, zin_b, KP0);
    lstm_rec_kernel<<<2, 800, 0, stream>>>(zin_f, zin_b, ph0f, ph0b,
                                           x2b, nullptr, nullptr, 0);
    // --- layer 1 ---
    gemm_zin_kernel<<<ggrid, 256, 0, stream>>>(x2b, pw1f, b1f, zin_f, KP1);
    gemm_zin_kernel<<<ggrid, 256, 0, stream>>>(x2b, pw1b, b1b, zin_b, KP1);
    lstm_rec_kernel<<<2, 800, 0, stream>>>(zin_f, zin_b, ph1f, ph1b,
                                           nullptr, hf_o, hb_o, 1);

    // --- outputs ---
    float* out = (float*)d_out;
    span_kernel<<<BB * NP, 256, 0, stream>>>(hf_o, hb_o, out);
    senlens_kernel<<<BB, 128, 0, stream>>>(pidx, out + (size_t)BB * NP * (2 * HH));
}
